// ConvLSTMLatentCell_66228395704765
// MI455X (gfx1250) — matrix operand order repaired and checked by reading, not compiled, not run on hardware
//
#include <hip/hip_runtime.h>

typedef __attribute__((ext_vector_type(16))) _Float16 v16h;
typedef __attribute__((ext_vector_type(8)))  _Float16 v8h;
typedef __attribute__((ext_vector_type(8)))  float    v8f;

#define NPIX   65536   // B*H*W = 16*64*64
#define BSZ    16
#define TSZ    8
#define CIN0   16
#define HID    32

__device__ __forceinline__ float sigmoid_fast(float x) {
    return 1.0f / (1.0f + __expf(-x));
}
__device__ __forceinline__ float tanh_fast(float x) {
    float e = __expf(2.0f * x);
    return 1.0f - 2.0f / (e + 1.0f);
}

// Gate-interleaved output-channel permutation: packed row mp (0..127) ->
// original cout.  Wave w = mp>>4 owns hid channels w*4..w*4+3 with row order
// [i0..3 | o0..3 | f0..3 | g0..3], so lanes 0-15 of the C/D fragment hold
// {i,o} and lanes 16-31 hold {f,g} for the same (hid, pixel).
__device__ __forceinline__ int gate_perm(int mp) {
    const int w = mp >> 4, idx = mp & 15;
    const int sel = idx >> 2, q = idx & 3;
    const int base = (sel == 0) ? 0 : (sel == 1) ? 64 : (sel == 2) ? 32 : 96;
    return base + (w << 2) + q;
}

// ---------------------------------------------------------------------------
// Prologue kernels
// ---------------------------------------------------------------------------

__global__ void zero_u32_kernel(unsigned* __restrict__ p, long long n) {
    long long i = (long long)blockIdx.x * blockDim.x + threadIdx.x;
    long long stride = (long long)gridDim.x * blockDim.x;
    for (; i < n; i += stride) p[i] = 0u;
}

// hidden = relu((-0.5 + midi/64) @ W1 + b1)   -> [B*T][4] f32
__global__ void pe_hidden_kernel(const float* __restrict__ midi,
                                 const float* __restrict__ w1,   // [2][4]
                                 const float* __restrict__ b1,   // [4]
                                 float* __restrict__ hid4) {
    int idx = blockIdx.x * blockDim.x + threadIdx.x;
    if (idx >= BSZ * TSZ * 4) return;
    int bt = idx >> 2, j = idx & 3;
    float n0 = -0.5f + midi[bt * 2 + 0] * (1.0f / 64.0f);
    float n1 = -0.5f + midi[bt * 2 + 1] * (1.0f / 64.0f);
    float v = n0 * w1[0 * 4 + j] + n1 * w1[1 * 4 + j] + b1[j];
    hid4[idx] = v > 0.0f ? v : 0.0f;
}

// xpe[t][c][b*4096 + sp] = f16( x_audio[b][t][c][sp] + tanh(hid4 . W2[:,peidx] + b2) )
__global__ void pe_apply_kernel(const float* __restrict__ x_audio,
                                const float* __restrict__ hid4,  // [B*T][4]
                                const float* __restrict__ w2,    // [4][65536]
                                const float* __restrict__ b2,    // [65536]
                                _Float16* __restrict__ xpe) {    // [T][16][NPIX]
    long long idx = (long long)blockIdx.x * blockDim.x + threadIdx.x;
    if (idx >= (long long)TSZ * CIN0 * NPIX) return;
    int pix = (int)(idx & (NPIX - 1));
    int c   = (int)((idx >> 16) & 15);
    int t   = (int)(idx >> 20);
    int b   = pix >> 12;
    int sp  = pix & 4095;                 // y*64 + x
    int peidx = c * 4096 + sp;
    const float* hp = hid4 + (b * TSZ + t) * 4;
    float pe = hp[0] * w2[0 * 65536 + peidx] + hp[1] * w2[1 * 65536 + peidx] +
               hp[2] * w2[2 * 65536 + peidx] + hp[3] * w2[3 * 65536 + peidx] + b2[peidx];
    pe = tanh_fast(pe);
    float xa = x_audio[((((long long)b * TSZ + t) * CIN0 + c) << 12) + sp];
    xpe[idx] = (_Float16)(xa + pe);
}

// OIHW f32 -> [9][COUT][CPAD] f16 (zero-padded channels, optional gate perm)
__global__ void pack_weights_kernel(const float* __restrict__ src,
                                    _Float16* __restrict__ dst,
                                    int COUT, int CIN, int CPAD, int permute) {
    int total = 9 * COUT * CPAD;
    int idx = blockIdx.x * blockDim.x + threadIdx.x;
    if (idx >= total) return;
    int c   = idx % CPAD;
    int o   = (idx / CPAD) % COUT;
    int off = idx / (CPAD * COUT);
    int osrc = permute ? gate_perm(o) : o;
    dst[idx] = (c < CIN) ? (_Float16)src[(osrc * CIN + c) * 9 + off] : (_Float16)0.0f;
}

__global__ void pack_bias_kernel(const float* __restrict__ src,
                                 float* __restrict__ dst) {
    int i = threadIdx.x;
    if (i < 128) dst[i] = src[gate_perm(i)];
}

// ---------------------------------------------------------------------------
// Fused implicit-GEMM conv (WMMA) + LSTM gates, no z spill:
//   gate-permuted cout packing => each wave holds all 4 gates for 4 hid
//   channels; one cross-half lane shuffle assembles (i,f,o,g) in-lane.
//   LDS: only the 66-column shifted B window (4.3 KB).
// ---------------------------------------------------------------------------
__global__ __launch_bounds__(256)
void conv_lstm_step_kernel(const _Float16* __restrict__ wpack, // [9][128][CPAD], gate-permuted
                           const float*    __restrict__ bias,  // [128], gate-permuted
                           const _Float16* __restrict__ srcA,  // [CA][NPIX]
                           int CA,
                           const _Float16* __restrict__ srcB,  // [32][NPIX] (h_{t-1})
                           int CPAD,
                           float*    __restrict__ c_state,     // [32][NPIX]
                           _Float16* __restrict__ h_out,       // [32][NPIX]
                           _Float16* __restrict__ seq_out) {   // optional [32][NPIX]
    __shared__ __align__(32) _Float16 ldsB[68 * 32];   // [col 0..65][32 k]

    const int tid    = threadIdx.x;
    const int lane   = tid & 31;
    const int wave   = tid >> 5;
    const int lanelo = lane & 15;
    const int hi     = lane >> 4;

    const int nb      = blockIdx.x;       // 0 .. B*H-1
    const int bb      = nb >> 6;
    const int yy0     = nb & 63;
    const int rowbase = bb * 4096 + yy0 * 64;

    // constant-zero W-padding edge columns (written once)
    if (tid < 8) {
        v8h z = {};
        const int col = (tid >> 2) ? 65 : 0;
        *(v8h*)(ldsB + col * 32 + (tid & 3) * 8) = z;
    }

    v8f acc0 = {}, acc1 = {}, acc2 = {}, acc3 = {};
    const int nchunk = CPAD >> 5;
    const int CB = HID;

    // staging decomposition: thread -> (pixel column, 8-channel group)
    // g == wave>>1 is wave-uniform: force scalar so src select is SALU.
    const int px = ((wave & 1) << 5) | lane;
    const int gs = __builtin_amdgcn_readfirstlane(wave >> 1);   // 0..3

    for (int ky = 0; ky < 3; ++ky) {
        const int ysrc = yy0 + ky - 1;
        const bool yok = (unsigned)ysrc < 64u;        // block-uniform
        const int srow = bb * 4096 + ysrc * 64;
        for (int ch = 0; ch < nchunk; ++ch) {
            const int kc = ch << 5;
            // ---- stage B window: cols 1..64 (xs = px), k = kc..kc+31 ----
            {
                const int c0 = kc + (gs << 3);        // scalar; aligned with CA / CA+CB
                const _Float16* base = nullptr;
                if (yok) {
                    if (c0 < CA)            base = srcA + (size_t)c0 * NPIX;
                    else if (c0 < CA + CB)  base = srcB + (size_t)(c0 - CA) * NPIX;
                }
                v8h v = {};
                if (base) {
                    const _Float16* sp = base + srow + px;
                    #pragma unroll
                    for (int k = 0; k < 8; ++k) v[k] = sp[(size_t)k * NPIX];
                }
                *(v8h*)(ldsB + (px + 1) * 32 + (gs << 3)) = v;
            }
            __syncthreads();
            // ---- 3 kx taps x 4 N-tiles of WMMA over this staged window ----
            #pragma unroll
            for (int kx = 0; kx < 3; ++kx) {
                const int off = ky * 3 + kx;
                const _Float16* pa =
                    wpack + ((size_t)off * 128 + wave * 16 + lanelo) * CPAD + kc;
                v8h alo = *(const v8h*)(pa + hi * 16);
                v8h ahi = *(const v8h*)(pa + hi * 16 + 8);
                v16h afrag;
                #pragma unroll
                for (int j = 0; j < 8; ++j) { afrag[j] = alo[j]; afrag[j + 8] = ahi[j]; }
                const _Float16* pb = ldsB + (lanelo + kx) * 32 + hi * 16;
                const v16h b0 = *(const v16h*)(pb);
                acc0 = __builtin_amdgcn_wmma_f32_16x16x32_f16(false, afrag, false, b0, (short)0, acc0, false, false);
                const v16h b1 = *(const v16h*)(pb + 16 * 32);
                acc1 = __builtin_amdgcn_wmma_f32_16x16x32_f16(false, afrag, false, b1, (short)0, acc1, false, false);
                const v16h b2 = *(const v16h*)(pb + 32 * 32);
                acc2 = __builtin_amdgcn_wmma_f32_16x16x32_f16(false, afrag, false, b2, (short)0, acc2, false, false);
                const v16h b3 = *(const v16h*)(pb + 48 * 32);
                acc3 = __builtin_amdgcn_wmma_f32_16x16x32_f16(false, afrag, false, b3, (short)0, acc3, false, false);
            }
            __syncthreads();
        }
    }

    // ---- in-register LSTM gates (no LDS z tile) ----
    // wave-uniform permuted bias slice
    float bw[16];
    #pragma unroll
    for (int j = 0; j < 16; ++j) bw[j] = bias[(wave << 4) + j];
    const int hw = wave << 2;                       // first hid channel of wave
    const bool has_seq = (seq_out != nullptr);      // block-uniform

    v8f accs[4] = { acc0, acc1, acc2, acc3 };
    #pragma unroll
    for (int nt = 0; nt < 4; ++nt) {
        const int pxn = nt * 16 + lanelo;
        float pf[8];
        #pragma unroll
        for (int j = 0; j < 8; ++j) pf[j] = __shfl_xor(accs[nt][j], 16, 32);
        if (hi == 0) {
            // own rows: i(hw+0..3), o(hw+0..3); partner: f, g.  Handle hid hw+0, hw+1.
            #pragma unroll
            for (int e = 0; e < 2; ++e) {
                const float iv = accs[nt][e]     + bw[e];
                const float ov = accs[nt][4 + e] + bw[4 + e];
                const float fv = pf[e]           + bw[8 + e];
                const float gv = pf[4 + e]       + bw[12 + e];
                const int hd = hw + e;
                const size_t gidx = (size_t)hd * NPIX + rowbase + pxn;
                const float cprev = c_state[gidx];
                const float c2 = sigmoid_fast(fv) * cprev + sigmoid_fast(iv) * tanh_fast(gv);
                const float h2 = sigmoid_fast(ov) * tanh_fast(c2);
                c_state[gidx] = c2;
                const _Float16 hh = (_Float16)h2;
                h_out[gidx] = hh;
                if (has_seq) seq_out[gidx] = hh;
            }
        } else {
            // own rows: f(hw+0..3), g(hw+0..3); partner: i, o.  Handle hid hw+2, hw+3.
            #pragma unroll
            for (int e = 0; e < 2; ++e) {
                const float fv = accs[nt][2 + e] + bw[10 + e];
                const float gv = accs[nt][6 + e] + bw[14 + e];
                const float iv = pf[2 + e]       + bw[2 + e];
                const float ov = pf[6 + e]       + bw[6 + e];
                const int hd = hw + 2 + e;
                const size_t gidx = (size_t)hd * NPIX + rowbase + pxn;
                const float cprev = c_state[gidx];
                const float c2 = sigmoid_fast(fv) * cprev + sigmoid_fast(iv) * tanh_fast(gv);
                const float h2 = sigmoid_fast(ov) * tanh_fast(c2);
                c_state[gidx] = c2;
                const _Float16 hh = (_Float16)h2;
                h_out[gidx] = hh;
                if (has_seq) seq_out[gidx] = hh;
            }
        }
    }
}

// ---------------------------------------------------------------------------
// Final 3x3 conv: Cin=64 -> Cout=64 (no permutation), output [B][64][H][W] f32
// ---------------------------------------------------------------------------
__global__ __launch_bounds__(128)
void final_conv_kernel(const _Float16* __restrict__ wpack,  // [9][64][64]
                       const float*    __restrict__ bias,   // [64]
                       const _Float16* __restrict__ src,    // [64][NPIX]
                       float* __restrict__ out) {           // [B][64][4096]
    __shared__ __align__(32) _Float16 ldsB[68 * 32];

    const int tid    = threadIdx.x;
    const int lane   = tid & 31;
    const int wave   = tid >> 5;
    const int lanelo = lane & 15;
    const int hi     = lane >> 4;

    const int nb  = blockIdx.x;
    const int bb  = nb >> 6;
    const int yy0 = nb & 63;

    if (tid < 8) {
        v8h z = {};
        const int col = (tid >> 2) ? 65 : 0;
        *(v8h*)(ldsB + col * 32 + (tid & 3) * 8) = z;
    }

    v8f acc0 = {}, acc1 = {}, acc2 = {}, acc3 = {};
    const int px = ((wave & 1) << 5) | lane;
    const int gs = __builtin_amdgcn_readfirstlane(wave >> 1);   // 0..1 -> 16 ch

    for (int ky = 0; ky < 3; ++ky) {
        const int ysrc = yy0 + ky - 1;
        const bool yok = (unsigned)ysrc < 64u;
        const int srow = bb * 4096 + ysrc * 64;
        for (int ch = 0; ch < 2; ++ch) {
            const int kc = ch << 5;
            {   // stage B window: 16 channels per thread -> 2x v8h stores
                const int c0 = kc + (gs << 4);
                v8h v0 = {}, v1 = {};
                if (yok) {
                    const _Float16* sp = src + (size_t)c0 * NPIX + srow + px;
                    #pragma unroll
                    for (int k = 0; k < 8; ++k) {
                        v0[k] = sp[(size_t)k * NPIX];
                        v1[k] = sp[(size_t)(k + 8) * NPIX];
                    }
                }
                *(v8h*)(ldsB + (px + 1) * 32 + (gs << 4))     = v0;
                *(v8h*)(ldsB + (px + 1) * 32 + (gs << 4) + 8) = v1;
            }
            __syncthreads();
            #pragma unroll
            for (int kx = 0; kx < 3; ++kx) {
                const int off = ky * 3 + kx;
                const _Float16* pa =
                    wpack + ((size_t)off * 64 + wave * 16 + lanelo) * 64 + kc;
                v8h alo = *(const v8h*)(pa + hi * 16);
                v8h ahi = *(const v8h*)(pa + hi * 16 + 8);
                v16h afrag;
                #pragma unroll
                for (int j = 0; j < 8; ++j) { afrag[j] = alo[j]; afrag[j + 8] = ahi[j]; }
                const _Float16* pb = ldsB + (lanelo + kx) * 32 + hi * 16;
                const v16h b0 = *(const v16h*)(pb);
                acc0 = __builtin_amdgcn_wmma_f32_16x16x32_f16(false, afrag, false, b0, (short)0, acc0, false, false);
                const v16h b1 = *(const v16h*)(pb + 16 * 32);
                acc1 = __builtin_amdgcn_wmma_f32_16x16x32_f16(false, afrag, false, b1, (short)0, acc1, false, false);
                const v16h b2 = *(const v16h*)(pb + 32 * 32);
                acc2 = __builtin_amdgcn_wmma_f32_16x16x32_f16(false, afrag, false, b2, (short)0, acc2, false, false);
                const v16h b3 = *(const v16h*)(pb + 48 * 32);
                acc3 = __builtin_amdgcn_wmma_f32_16x16x32_f16(false, afrag, false, b3, (short)0, acc3, false, false);
            }
            __syncthreads();
        }
    }
    // store [B][cout][y][x]
    #pragma unroll
    for (int r = 0; r < 8; ++r) {
        const int m = wave * 16 + r + hi * 8;
        const float bm = bias[m];
        float* o = out + ((size_t)bb * 64 + m) * 4096 + yy0 * 64;
        o[ 0 + lanelo] = acc0[r] + bm;
        o[16 + lanelo] = acc1[r] + bm;
        o[32 + lanelo] = acc2[r] + bm;
        o[48 + lanelo] = acc3[r] + bm;
    }
}

// ---------------------------------------------------------------------------
// Host orchestration
// ---------------------------------------------------------------------------
extern "C" void kernel_launch(void* const* d_in, const int* in_sizes, int n_in,
                              void* d_out, int out_size, void* d_ws, size_t ws_size,
                              hipStream_t stream) {
    (void)in_sizes; (void)n_in; (void)out_size; (void)ws_size;
    const float* x_audio = (const float*)d_in[0];
    const float* midi    = (const float*)d_in[1];
    const float* ew1 = (const float*)d_in[2];
    const float* eb1 = (const float*)d_in[3];
    const float* ew2 = (const float*)d_in[4];
    const float* eb2 = (const float*)d_in[5];
    const float* w0f = (const float*)d_in[6];
    const float* b0f = (const float*)d_in[7];
    const float* w0b = (const float*)d_in[8];
    const float* b0b = (const float*)d_in[9];
    const float* w1f = (const float*)d_in[10];
    const float* b1f = (const float*)d_in[11];
    const float* w1b = (const float*)d_in[12];
    const float* b1b = (const float*)d_in[13];
    const float* fcw = (const float*)d_in[14];
    const float* fcb = (const float*)d_in[15];

    char* ws = (char*)d_ws;
    size_t o = 0;
    auto take = [&](size_t bytes) -> size_t {
        size_t r = o; o = (o + bytes + 255) & ~(size_t)255; return r;
    };
    _Float16* wp0f  = (_Float16*)(ws + take((size_t)9 * 128 * 64 * 2));
    _Float16* wp0b  = (_Float16*)(ws + take((size_t)9 * 128 * 64 * 2));
    _Float16* wp1f  = (_Float16*)(ws + take((size_t)9 * 128 * 96 * 2));
    _Float16* wp1b  = (_Float16*)(ws + take((size_t)9 * 128 * 96 * 2));
    _Float16* wpfc  = (_Float16*)(ws + take((size_t)9 * 64 * 64 * 2));
    float*    bp0f  = (float*)   (ws + take(128 * 4));
    float*    bp0b  = (float*)   (ws + take(128 * 4));
    float*    bp1f  = (float*)   (ws + take(128 * 4));
    float*    bp1b  = (float*)   (ws + take(128 * 4));
    _Float16* xpe   = (_Float16*)(ws + take((size_t)TSZ * CIN0 * NPIX * 2));
    _Float16* hseq0 = (_Float16*)(ws + take((size_t)TSZ * 64 * NPIX * 2));
    _Float16* hfin  = (_Float16*)(ws + take((size_t)64 * NPIX * 2));
    size_t state_off = take((size_t)HID * NPIX * 2);              // h16[0]
    _Float16* h16a = (_Float16*)(ws + state_off);
    _Float16* h16b = (_Float16*)(ws + take((size_t)HID * NPIX * 2));
    float*    cst  = (float*)   (ws + take((size_t)HID * NPIX * 4));
    float*    phid = (float*)   (ws + take((size_t)BSZ * TSZ * 4 * 4));
    _Float16* h16[2] = { h16a, h16b };

    // prologue: PE + weight/bias packing (gate-permuted for LSTM layers)
    pe_hidden_kernel<<<2, 256, 0, stream>>>(midi, ew1, eb1, phid);
    pe_apply_kernel<<<(TSZ * CIN0 * NPIX) / 256, 256, 0, stream>>>(x_audio, phid, ew2, eb2, xpe);
    pack_weights_kernel<<<(9 * 128 * 64 + 255) / 256, 256, 0, stream>>>(w0f, wp0f, 128, 48, 64, 1);
    pack_weights_kernel<<<(9 * 128 * 64 + 255) / 256, 256, 0, stream>>>(w0b, wp0b, 128, 48, 64, 1);
    pack_weights_kernel<<<(9 * 128 * 96 + 255) / 256, 256, 0, stream>>>(w1f, wp1f, 128, 96, 96, 1);
    pack_weights_kernel<<<(9 * 128 * 96 + 255) / 256, 256, 0, stream>>>(w1b, wp1b, 128, 96, 96, 1);
    pack_weights_kernel<<<(9 * 64 * 64 + 255) / 256, 256, 0, stream>>>(fcw, wpfc, 64, 64, 64, 0);
    pack_bias_kernel<<<1, 128, 0, stream>>>(b0f, bp0f);
    pack_bias_kernel<<<1, 128, 0, stream>>>(b0b, bp0b);
    pack_bias_kernel<<<1, 128, 0, stream>>>(b1f, bp1f);
    pack_bias_kernel<<<1, 128, 0, stream>>>(b1b, bp1b);

    // zero h16a + h16b + c_state (contiguous)
    const long long zwords = ((long long)HID * NPIX * 2 * 2 + (long long)HID * NPIX * 4) / 4;
    auto zero_state = [&]() {
        zero_u32_kernel<<<2048, 256, 0, stream>>>((unsigned*)(ws + state_off), zwords);
    };
    const int NB = BSZ * 64;   // 1024 blocks (one per image row)

    // layer 0 forward
    zero_state();
    for (int t = 0; t < TSZ; ++t)
        conv_lstm_step_kernel<<<NB, 256, 0, stream>>>(
            wp0f, bp0f, xpe + (size_t)t * CIN0 * NPIX, CIN0, h16[t & 1], 64,
            cst, h16[(t + 1) & 1], hseq0 + ((size_t)t * 64 + 0) * NPIX);

    // layer 0 backward
    zero_state();
    for (int s = 0; s < TSZ; ++s) {
        int t = TSZ - 1 - s;
        conv_lstm_step_kernel<<<NB, 256, 0, stream>>>(
            wp0b, bp0b, xpe + (size_t)t * CIN0 * NPIX, CIN0, h16[s & 1], 64,
            cst, h16[(s + 1) & 1], hseq0 + ((size_t)t * 64 + HID) * NPIX);
    }

    // layer 1 forward (only h at t=T-1 is ever consumed downstream)
    zero_state();
    for (int t = 0; t < TSZ; ++t)
        conv_lstm_step_kernel<<<NB, 256, 0, stream>>>(
            wp1f, bp1f, hseq0 + (size_t)t * 64 * NPIX, 64, h16[t & 1], 96,
            cst, h16[(t + 1) & 1], (t == TSZ - 1) ? hfin : (_Float16*)nullptr);

    // layer 1 backward: flip(scan(flip(x)))[-1] == first step of backward scan
    zero_state();
    conv_lstm_step_kernel<<<NB, 256, 0, stream>>>(
        wp1b, bp1b, hseq0 + (size_t)(TSZ - 1) * 64 * NPIX, 64, h16[0], 96,
        cst, h16[1], hfin + (size_t)HID * NPIX);

    // final conv -> d_out [B][64][H][W] f32
    final_conv_kernel<<<NB, 128, 0, stream>>>(wpfc, fcb, hfin, (float*)d_out);
}
